// OSae_33294586479265
// MI455X (gfx1250) — matrix operand order repaired and checked by reading, not compiled, not run on hardware
//
#include <hip/hip_runtime.h>
#include <hip/hip_bf16.h>

#define K_GRP   32
#define G_FEAT  2048
#define D_DIM   2048
#define B_BATCH 2048

#define TM 128
#define TN 128
#define TK 64
#define LSTR 72   // padded LDS row stride (bf16 elems): 144B, 16B-aligned, conflict-free

#define USE_ASYNC_A 1   // global_load_async_to_lds_b128 for the A tile

typedef __attribute__((ext_vector_type(16))) __bf16 v16bf;
typedef __attribute__((ext_vector_type(8)))  float  v8f;

__device__ __forceinline__ unsigned f2ord(float f) {
  unsigned u = __float_as_uint(f);
  return (u & 0x80000000u) ? ~u : (u | 0x80000000u);
}
__device__ __forceinline__ float ord2f(unsigned o) {
  unsigned u = (o & 0x80000000u) ? (o & 0x7FFFFFFFu) : ~o;
  return __uint_as_float(u);
}

#if USE_ASYNC_A
// Async DMA: 16B global -> LDS, tracked by ASYNCcnt (ISA 15.18.3 op 98).
__device__ __forceinline__ void async_copy_b128(unsigned ldsAddr, unsigned long long gAddr) {
  asm volatile("global_load_async_to_lds_b128 %0, %1, off"
               :: "v"(ldsAddr), "v"(gAddr) : "memory");
}
__device__ __forceinline__ void wait_async0() {
  asm volatile("s_wait_asynccnt 0x0" ::: "memory");
}
#endif

// ---------------- pass 0: zero accumulators ----------------
__global__ void sae_init(unsigned long long* __restrict__ maxbuf,
                         float* __restrict__ colsum, float* __restrict__ colsumsq,
                         float* __restrict__ e_accum, int* __restrict__ counts) {
  int i = blockIdx.x * blockDim.x + threadIdx.x;   // 65536 threads
  maxbuf[i] = 0ull;
  counts[i] = 0;
  if (i < D_DIM) { colsum[i] = 0.f; colsumsq[i] = 0.f; }
  if (i == 0) e_accum[0] = 0.f;
}

// ---------------- pass 1: xc = x - b_dec -> bf16, column sums ----------------
__global__ __launch_bounds__(256) void sae_prep(
    const float* __restrict__ x, const float* __restrict__ b_dec,
    __bf16* __restrict__ xcb, float* __restrict__ colsum, float* __restrict__ colsumsq) {
  int d  = blockIdx.x * 256 + threadIdx.x;   // gridDim.x = 8
  int b0 = blockIdx.y * 256;                 // gridDim.y = 8
  float bd = b_dec[d];
  float s = 0.f, s2 = 0.f;
  for (int b = b0; b < b0 + 256; ++b) {
    float v = x[(size_t)b * D_DIM + d] - bd;
    s += v; s2 += v * v;
    xcb[(size_t)b * D_DIM + d] = (__bf16)v;
  }
  atomicAdd(&colsum[d], s);
  atomicAdd(&colsumsq[d], s2);
}

// ---------------- pass 2: fused GEMM + per-group max/argmax ----------------
__global__ __launch_bounds__(256, 1) void sae_gemm_argmax(
    const __bf16* __restrict__ xcb, const float* __restrict__ enc,
    const float* __restrict__ b_enc, unsigned long long* __restrict__ maxbuf) {
  __shared__ __bf16 lA[2][TM * LSTR];
  __shared__ __bf16 lB[2][TN * LSTR];

  const int t    = threadIdx.x;
  const int wave = t >> 5, lane = t & 31;
  const int l15  = lane & 15, half = lane >> 4;
  const int wm   = wave & 3;      // 4 M slots x 32 rows
  const int wn   = wave >> 2;     // 2 N slots x 64 cols
  const int mTile = blockIdx.x;   // 16 tiles  (fast dim -> x stays L2 hot)
  const int nTile = blockIdx.y;   // 512 tiles
  const int mBase = mTile * TM;
  const size_t fBase = (size_t)nTile * TN;

  v8f acc[2][4];
#pragma unroll
  for (int mi = 0; mi < 2; ++mi)
#pragma unroll
    for (int ni = 0; ni < 4; ++ni)
#pragma unroll
      for (int e = 0; e < 8; ++e) acc[mi][ni][e] = 0.f;

  // cooperative staging: each thread owns 32 contiguous elems of one row
  const int srow  = t >> 1;       // 0..127
  const int shalf = t & 1;        // 0..1
  const __bf16* aSrc = xcb + (size_t)(mBase + srow) * D_DIM + shalf * 32;
  const float*  bSrc = enc + (fBase + srow) * D_DIM + shalf * 32;

  union Stage { __bf16 h[32]; uint4 q[4]; };
  Stage sb;

  // B chunk: fp32 -> bf16 at load time (16 live VGPRs instead of 32)
  auto loadB = [&](int kc) {
    const float4* bp4 = (const float4*)(bSrc + kc);
#pragma unroll
    for (int j = 0; j < 8; ++j) {
      float4 f = bp4[j];
      sb.h[4 * j + 0] = (__bf16)f.x;
      sb.h[4 * j + 1] = (__bf16)f.y;
      sb.h[4 * j + 2] = (__bf16)f.z;
      sb.h[4 * j + 3] = (__bf16)f.w;
    }
  };

#if USE_ASYNC_A
  // A tile is bf16 already: DMA it straight into LDS, no VGPR transit.
  auto issueA = [&](int nb, int kc) {
    unsigned dst = (unsigned)(size_t)&lA[nb][srow * LSTR + shalf * 32];
    unsigned long long src = (unsigned long long)(size_t)(aSrc + kc);
#pragma unroll
    for (int j = 0; j < 4; ++j) async_copy_b128(dst + 16u * j, src + 16ull * j);
  };
  issueA(0, 0);
#else
  Stage sa;
  auto loadA = [&](int kc) {
    const uint4* ap4 = (const uint4*)(aSrc + kc);
#pragma unroll
    for (int j = 0; j < 4; ++j) sa.q[j] = ap4[j];
  };
  loadA(0);
#endif
  loadB(0);

  int buf = 0;
  for (int kc = 0; kc < D_DIM; kc += TK, buf ^= 1) {
    // publish staged B chunk to this iteration's LDS buffer
    uint4* bD = (uint4*)&lB[buf][srow * LSTR + shalf * 32];
#pragma unroll
    for (int j = 0; j < 4; ++j) bD[j] = sb.q[j];
#if USE_ASYNC_A
    wait_async0();               // our async A-writes into lA[buf] have landed
#else
    uint4* aD = (uint4*)&lA[buf][srow * LSTR + shalf * 32];
#pragma unroll
    for (int j = 0; j < 4; ++j) aD[j] = sa.q[j];
#endif
    __syncthreads();             // all waves' LDS writes visible

    // start fetching the next chunk (overlaps the 16 WMMAs below);
    // lA[buf^1]/lB[buf^1] were fully consumed before the barrier above.
    if (kc + TK < D_DIM) {
#if USE_ASYNC_A
      issueA(buf ^ 1, kc + TK);
#else
      loadA(kc + TK);
#endif
      loadB(kc + TK);
      if (kc + 2 * TK < D_DIM)
        __builtin_prefetch(bSrc + kc + 2 * TK, 0, 1);   // global_prefetch_b8
    }

    // consume: two K=32 sub-chunks
#pragma unroll
    for (int kk = 0; kk < 2; ++kk) {
      union Frag { v16bf v; uint4 q[2]; } af[2], bfr[4];
#pragma unroll
      for (int mi = 0; mi < 2; ++mi) {
        // 16-bit A 16x32 layout: lanes<16 hold K0-7 & K16-23, lanes>=16 hold K8-15 & K24-31
        const __bf16* ap = &lA[buf][(wm * 32 + mi * 16 + l15) * LSTR + kk * 32];
        af[mi].q[0] = *(const uint4*)(ap + half * 8);
        af[mi].q[1] = *(const uint4*)(ap + 16 + half * 8);
      }
#pragma unroll
      for (int ni = 0; ni < 4; ++ni) {
        const __bf16* bp = &lB[buf][(wn * 64 + ni * 16 + l15) * LSTR + kk * 32];
        bfr[ni].q[0] = *(const uint4*)(bp + half * 8);
        bfr[ni].q[1] = *(const uint4*)(bp + 16 + half * 8);
      }
#pragma unroll
      for (int mi = 0; mi < 2; ++mi)
#pragma unroll
        for (int ni = 0; ni < 4; ++ni)
          acc[mi][ni] = __builtin_amdgcn_wmma_f32_16x16x32_bf16(
              false, af[mi].v, false, bfr[ni].v, (short)0, acc[mi][ni], false, false);
    }
  }

  // epilogue: + b_enc, max/argmax over this block's 128 features, publish u64 key
  const int kIdx  = nTile >> 4;          // feature tile -> group (128 | G)
  const int gBase = (nTile & 15) * 128;
  float be[4];
#pragma unroll
  for (int ni = 0; ni < 4; ++ni)
    be[ni] = b_enc[kIdx * G_FEAT + gBase + wn * 64 + ni * 16 + l15];

#pragma unroll
  for (int mi = 0; mi < 2; ++mi) {
#pragma unroll
    for (int r = 0; r < 8; ++r) {
      float v = acc[mi][0][r] + be[0];
      int   c = wn * 64 + l15;
#pragma unroll
      for (int ni = 1; ni < 4; ++ni) {
        float tv = acc[mi][ni][r] + be[ni];
        int   tc = wn * 64 + ni * 16 + l15;
        if (tv > v || (tv == v && tc < c)) { v = tv; c = tc; }
      }
#pragma unroll
      for (int off = 1; off < 16; off <<= 1) {   // butterfly within 16-lane half
        float ov = __shfl_xor(v, off, 32);
        int   oc = __shfl_xor(c, off, 32);
        if (ov > v || (ov == v && oc < c)) { v = ov; c = oc; }
      }
      if (l15 == 0) {                            // lanes 0 & 16: rows r and r+8
        int row = mBase + wm * 32 + mi * 16 + half * 8 + r;
        int g   = gBase + c;
        unsigned long long key = ((unsigned long long)f2ord(v) << 32)
                               | (unsigned)(0x7FFFFFFF - g);  // ties -> lowest g
        atomicMax(&maxbuf[(size_t)row * K_GRP + kIdx], key);
      }
    }
  }
}

// ---------------- pass 3: feature histogram ----------------
__global__ void sae_hist(const unsigned long long* __restrict__ maxbuf,
                         int* __restrict__ counts) {
  int i = blockIdx.x * blockDim.x + threadIdx.x;   // B*K = 65536
  unsigned long long key = maxbuf[i];
  int k = i & (K_GRP - 1);
  int g = 0x7FFFFFFF - (int)(unsigned)(key & 0xFFFFFFFFu);
  atomicAdd(&counts[k * G_FEAT + g], 1);
}

// ---------------- pass 4: reconstruction + squared error ----------------
__global__ __launch_bounds__(256) void sae_decode(
    const float* __restrict__ x, const float* __restrict__ decoder,
    const float* __restrict__ b_dec, const unsigned long long* __restrict__ maxbuf,
    float* __restrict__ e_accum) {
  __shared__ float sval[K_GRP];
  __shared__ int   sg[K_GRP];
  __shared__ float red[256];
  const int b = blockIdx.x, t = threadIdx.x;
  if (t < K_GRP) {
    unsigned long long key = maxbuf[(size_t)b * K_GRP + t];
    sval[t] = ord2f((unsigned)(key >> 32));
    sg[t]   = 0x7FFFFFFF - (int)(unsigned)(key & 0xFFFFFFFFu);
  }
  __syncthreads();
  float local = 0.f;
#pragma unroll
  for (int j = 0; j < D_DIM / 256; ++j) {
    int d = t + j * 256;
    float out = b_dec[d];
#pragma unroll 4
    for (int k = 0; k < K_GRP; ++k)
      out += sval[k] * decoder[((size_t)k * G_FEAT + sg[k]) * D_DIM + d];
    float xc = x[(size_t)b * D_DIM + d] - b_dec[d];
    float diff = xc - out;
    local += diff * diff;
  }
  red[t] = local; __syncthreads();
  for (int s = 128; s > 0; s >>= 1) { if (t < s) red[t] += red[t + s]; __syncthreads(); }
  if (t == 0) atomicAdd(e_accum, red[0]);
}

// ---------------- pass 5: finalize fvu ----------------
__global__ __launch_bounds__(256) void sae_fvu(
    const float* __restrict__ colsum, const float* __restrict__ colsumsq,
    const float* __restrict__ e_accum, float* __restrict__ out) {
  __shared__ float red[256];
  const int t = threadIdx.x;
  float tv = 0.f;
  for (int d = t; d < D_DIM; d += 256) {
    float s = colsum[d];
    tv += colsumsq[d] - s * s * (1.0f / (float)B_BATCH);
  }
  red[t] = tv; __syncthreads();
  for (int s = 128; s > 0; s >>= 1) { if (t < s) red[t] += red[t + s]; __syncthreads(); }
  if (t == 0) {
    out[0] = e_accum[0] / red[0];            // fvu
    out[1 + K_GRP * G_FEAT] = 0.0f;          // aux_fvu
  }
}

extern "C" void kernel_launch(void* const* d_in, const int* in_sizes, int n_in,
                              void* d_out, int out_size, void* d_ws, size_t ws_size,
                              hipStream_t stream) {
  const float* x       = (const float*)d_in[0];
  const float* encoder = (const float*)d_in[1];
  const float* b_enc   = (const float*)d_in[2];
  const float* decoder = (const float*)d_in[3];
  const float* b_dec   = (const float*)d_in[4];

  float* out    = (float*)d_out;
  int*   counts = ((int*)d_out) + 1;   // out layout: [fvu][65536 counts][aux_fvu]

  char* ws = (char*)d_ws;
  __bf16*             xcb      = (__bf16*)ws;                              // 8 MB
  unsigned long long* maxbuf   = (unsigned long long*)(ws + 8388608);      // 512 KB
  float*              colsum   = (float*)(ws + 8912896);                   // 8 KB
  float*              colsumsq = (float*)(ws + 8921088);                   // 8 KB
  float*              e_accum  = (float*)(ws + 8929280);                   // 4 B

  sae_init<<<256, 256, 0, stream>>>(maxbuf, colsum, colsumsq, e_accum, counts);
  sae_prep<<<dim3(D_DIM / 256, 8), 256, 0, stream>>>(x, b_dec, xcb, colsum, colsumsq);
  sae_gemm_argmax<<<dim3(B_BATCH / TM, (K_GRP * G_FEAT) / TN), 256, 0, stream>>>(
      xcb, encoder, b_enc, maxbuf);
  sae_hist<<<(B_BATCH * K_GRP) / 256, 256, 0, stream>>>(maxbuf, counts);
  sae_decode<<<B_BATCH, 256, 0, stream>>>(x, decoder, b_dec, maxbuf, e_accum);
  sae_fvu<<<1, 256, 0, stream>>>(colsum, colsumsq, e_accum, out);
}
